// NRConvBlock_16870631539379
// MI455X (gfx1250) — matrix operand order repaired and checked by reading, not compiled, not run on hardware
//
#include <hip/hip_runtime.h>

typedef __attribute__((ext_vector_type(16))) __bf16 v16bf;
typedef __attribute__((ext_vector_type(8)))  __bf16 v8bf;
typedef __attribute__((ext_vector_type(8)))  float  v8f;

#define BN_EPS 1e-3f

// ---------------------------------------------------------------------------
// WMMA wrapper: D = A(16x32 bf16) * B(32x16 bf16) + C(16x16 f32)
// ---------------------------------------------------------------------------
__device__ __forceinline__ v8f wmma_bf16(v16bf a, v16bf b, v8f c) {
    return __builtin_amdgcn_wmma_f32_16x16x32_bf16(
        /*neg_a=*/false, a, /*neg_b=*/false, b,
        /*c_mod=*/(short)0, c, /*reuse_a=*/false, /*reuse_b=*/false);
}

// ---------------------------------------------------------------------------
// A-fragment loaders. 16-bit A 16x32 layout (ISA 7.12.2):
//   lanes 0-15  (M=lane)    hold K = 0..7  (elem 0..7)  and K = 16..23 (elem 8..15)
//   lanes 16-31 (M=lane-16) hold K = 8..15 (elem 0..7)  and K = 24..31 (elem 8..15)
// => per lane: 8 contiguous channels at c0 and 8 at c0+16, c0 = (lane>>4)*8.
// ---------------------------------------------------------------------------
__device__ __forceinline__ v16bf load_a_f32(const float* __restrict__ row) {
    const float4 x0 = *reinterpret_cast<const float4*>(row);
    const float4 x1 = *reinterpret_cast<const float4*>(row + 4);
    const float4 x2 = *reinterpret_cast<const float4*>(row + 16);
    const float4 x3 = *reinterpret_cast<const float4*>(row + 20);
    v16bf a;
    a[0]  = (__bf16)x0.x; a[1]  = (__bf16)x0.y; a[2]  = (__bf16)x0.z; a[3]  = (__bf16)x0.w;
    a[4]  = (__bf16)x1.x; a[5]  = (__bf16)x1.y; a[6]  = (__bf16)x1.z; a[7]  = (__bf16)x1.w;
    a[8]  = (__bf16)x2.x; a[9]  = (__bf16)x2.y; a[10] = (__bf16)x2.z; a[11] = (__bf16)x2.w;
    a[12] = (__bf16)x3.x; a[13] = (__bf16)x3.y; a[14] = (__bf16)x3.z; a[15] = (__bf16)x3.w;
    return a;
}

__device__ __forceinline__ v16bf load_a_bf16(const __bf16* __restrict__ row) {
    v8bf lo = *reinterpret_cast<const v8bf*>(row);
    v8bf hi = *reinterpret_cast<const v8bf*>(row + 16);
    v16bf a;
#pragma unroll
    for (int e = 0; e < 8; ++e) { a[e] = lo[e]; a[8 + e] = hi[e]; }
    return a;
}

// ---------------------------------------------------------------------------
// Weight -> B-fragment conversion (f32 -> bf16).
// 16-bit B 32x16 layout (matches A): lanes 0-15 hold K=0..7 (e 0..7) and
// K=16..23 (e 8..15), lanes 16-31 hold K=8..15 and K=24..31.
//   frag[j][lane][e] = W[tap][ (lane>>4)*8 + (e&7) + (e>>3)*16 ][ coloff + j*16 + (lane&15) ]
// Buffer layout: [tap][j][lane][e] contiguous, 512 bf16 per fragment.
// ---------------------------------------------------------------------------
__global__ void prep_frag_kernel(const float* __restrict__ W, __bf16* __restrict__ dst,
                                 int taps, int ncols, int coloff) {
    int t = blockIdx.x * blockDim.x + threadIdx.x;
    int total = taps * 1024;
    if (t >= total) return;
    int frag = t >> 9;          // (tap, j)
    int rem  = t & 511;
    int lane = rem >> 4;
    int e    = rem & 15;
    int tap  = frag >> 1;
    int j    = frag & 1;
    int K    = ((lane >> 4) << 3) + (e & 7) + ((e >> 3) << 4);
    float v  = W[((size_t)tap * 32 + K) * ncols + coloff + j * 16 + (lane & 15)];
    dst[t] = (__bf16)v;
}

// ---------------------------------------------------------------------------
// Fused submanifold conv (gather + GEMM over KT taps) + BN + ReLU.
// One wave == TWO 16-point tiles (4 independent WMMA chains per tap, each
// B-fragment LDS read reused twice). Grid-stride over tile pairs.
// Output: bf16 rows (stride 32) always; optionally also f32 rows (stride 64).
// ---------------------------------------------------------------------------
template <bool IN_F32, int KT, bool DUAL>
__global__ __launch_bounds__(256) void subm_conv(
    const void* __restrict__ xin, int xstride,
    const int* __restrict__ nbr, const unsigned char* __restrict__ mask,
    const __bf16* __restrict__ wfrag,
    const float* __restrict__ bng, const float* __restrict__ bnb,
    const float* __restrict__ bnm, const float* __restrict__ bnv,
    __bf16* __restrict__ outbf, float* __restrict__ outf32, int npts) {
    __shared__ __bf16 sw[KT * 1024];
    {
        const unsigned int* s = (const unsigned int*)wfrag;
        unsigned int* d = (unsigned int*)sw;
        for (int i = threadIdx.x; i < KT * 512; i += blockDim.x) d[i] = s[i];
    }
    __syncthreads();

    const int lane = threadIdx.x & 31;
    const int half = lane >> 4;
    const int col  = lane & 15;
    const int c0   = half << 3;
    const int colB = col + 16;

    // BN fold: y = x*s + t
    const float sA = bng[col]  * rsqrtf(bnv[col]  + BN_EPS);
    const float tA = bnb[col]  - bnm[col]  * sA;
    const float sB = bng[colB] * rsqrtf(bnv[colB] + BN_EPS);
    const float tB = bnb[colB] - bnm[colB] * sB;

    const int ntiles  = (npts + 15) >> 4;
    const int npairs  = (ntiles + 1) >> 1;
    const int wid     = blockIdx.x * (blockDim.x >> 5) + (threadIdx.x >> 5);
    const int wstride = gridDim.x * (blockDim.x >> 5);

    for (int pair = wid; pair < npairs; pair += wstride) {
        const int tile0 = pair * 2;
        const int pr0   = tile0 * 16 + col;          // A-row points
        const int pr1   = pr0 + 16;
        const bool prv0 = pr0 < npts;
        const bool prv1 = pr1 < npts;
        const int prc0  = prv0 ? pr0 : npts - 1;
        const int prc1  = prv1 ? pr1 : npts - 1;
        v8f acc00 = {}, acc01 = {}, acc10 = {}, acc11 = {};
        for (int k = 0; k < KT; ++k) {
            const int idx0 = nbr[(size_t)prc0 * KT + k];
            const int idx1 = nbr[(size_t)prc1 * KT + k];
            const bool mv0 = prv0 && (mask[(size_t)prc0 * KT + k] != 0);
            const bool mv1 = prv1 && (mask[(size_t)prc1 * KT + k] != 0);
            v16bf a0 = {}, a1 = {};
            if (mv0) {
                if constexpr (IN_F32)
                    a0 = load_a_f32((const float*)xin + (size_t)idx0 * xstride + c0);
                else
                    a0 = load_a_bf16((const __bf16*)xin + (size_t)idx0 * xstride + c0);
            }
            if (mv1) {
                if constexpr (IN_F32)
                    a1 = load_a_f32((const float*)xin + (size_t)idx1 * xstride + c0);
                else
                    a1 = load_a_bf16((const __bf16*)xin + (size_t)idx1 * xstride + c0);
            }
            const v16bf b0 = *reinterpret_cast<const v16bf*>(&sw[(k * 2 + 0) * 512 + lane * 16]);
            const v16bf b1 = *reinterpret_cast<const v16bf*>(&sw[(k * 2 + 1) * 512 + lane * 16]);
            acc00 = wmma_bf16(a0, b0, acc00);
            acc01 = wmma_bf16(a0, b1, acc01);
            acc10 = wmma_bf16(a1, b0, acc10);
            acc11 = wmma_bf16(a1, b1, acc11);
        }
        // C/D layout: elem r -> point row (r + 8*half), column = lane&15 (+16)
#pragma unroll
        for (int r = 0; r < 8; ++r) {
            const int pt0 = tile0 * 16 + r + (half << 3);
            const int pt1 = pt0 + 16;
            if (pt0 < npts) {
                const float y0 = fmaxf(fmaf(acc00[r], sA, tA), 0.f);
                const float y1 = fmaxf(fmaf(acc01[r], sB, tB), 0.f);
                __bf16* ob = outbf + (size_t)pt0 * 32;
                ob[col] = (__bf16)y0; ob[colB] = (__bf16)y1;
                if constexpr (DUAL) {
                    float* of = outf32 + (size_t)pt0 * 64;
                    of[col] = y0; of[colB] = y1;
                }
            }
            if (pt1 < npts) {
                const float y0 = fmaxf(fmaf(acc10[r], sA, tA), 0.f);
                const float y1 = fmaxf(fmaf(acc11[r], sB, tB), 0.f);
                __bf16* ob = outbf + (size_t)pt1 * 32;
                ob[col] = (__bf16)y0; ob[colB] = (__bf16)y1;
                if constexpr (DUAL) {
                    float* of = outf32 + (size_t)pt1 * 64;
                    of[col] = y0; of[colB] = y1;
                }
            }
        }
    }
}

// ---------------------------------------------------------------------------
// Per-point multi-head cross attention + residual fuse.
// out[:,0:32] (=f3d) += ((softmax_h(q.k/sqrt8) * v).reshape(N,32)) @ wproj
// Reads f3d/f2d from bf16 shadow copies (no converts, half the bytes).
// frags: [q0,q1,k0,k1,v0,v1,p0,p1] * 512 bf16.
// ---------------------------------------------------------------------------
__global__ __launch_bounds__(256) void attn_fuse(float* out,
                                                 const __bf16* __restrict__ frags,
                                                 const __bf16* __restrict__ f3db,
                                                 const __bf16* __restrict__ f2db,
                                                 int npts) {
    __shared__ __bf16 sf[8 * 512];
    __shared__ float  swb[8 * 512];   // per-wave 16x32 scratch tile
    {
        const unsigned int* s = (const unsigned int*)frags;
        unsigned int* d = (unsigned int*)sf;
        for (int i = threadIdx.x; i < 2048; i += blockDim.x) d[i] = s[i];
    }
    __syncthreads();

    const int lane = threadIdx.x & 31;
    const int wloc = threadIdx.x >> 5;
    const int half = lane >> 4;
    const int col  = lane & 15;
    const int c0   = half << 3;
    const int hbit = (lane >> 3) & 1;   // which head within this acc's column group

    const v16bf qf0 = *reinterpret_cast<const v16bf*>(sf + 0 * 512 + lane * 16);
    const v16bf qf1 = *reinterpret_cast<const v16bf*>(sf + 1 * 512 + lane * 16);
    const v16bf kf0 = *reinterpret_cast<const v16bf*>(sf + 2 * 512 + lane * 16);
    const v16bf kf1 = *reinterpret_cast<const v16bf*>(sf + 3 * 512 + lane * 16);
    const v16bf vf0 = *reinterpret_cast<const v16bf*>(sf + 4 * 512 + lane * 16);
    const v16bf vf1 = *reinterpret_cast<const v16bf*>(sf + 5 * 512 + lane * 16);
    const v16bf pf0 = *reinterpret_cast<const v16bf*>(sf + 6 * 512 + lane * 16);
    const v16bf pf1 = *reinterpret_cast<const v16bf*>(sf + 7 * 512 + lane * 16);

    const int ntiles  = (npts + 15) >> 4;
    const int wid     = blockIdx.x * (blockDim.x >> 5) + (threadIdx.x >> 5);
    const int wstride = gridDim.x * (blockDim.x >> 5);
    const float inv_sqrt_dk = 0.3535533905932738f;   // 1/sqrt(8)

    for (int tile = wid; tile < ntiles; tile += wstride) {
        const int pr  = tile * 16 + col;
        const int prc = (pr < npts) ? pr : npts - 1;
        const v16bf a3 = load_a_bf16(f3db + (size_t)prc * 32 + c0);
        const v16bf a2 = load_a_bf16(f2db + (size_t)prc * 32 + c0);

        const v8f z = {};
        const v8f q0  = wmma_bf16(a3, qf0, z);
        const v8f q1  = wmma_bf16(a3, qf1, z);
        const v8f kk0 = wmma_bf16(a2, kf0, z);
        const v8f kk1 = wmma_bf16(a2, kf1, z);
        const v8f vv0 = wmma_bf16(a2, vf0, z);
        const v8f vv1 = wmma_bf16(a2, vf1, z);

        v8f w0, w1;
#pragma unroll
        for (int r = 0; r < 8; ++r) {
            // per-channel q*k products; reduce over 8-lane head groups
            float p0 = q0[r] * kk0[r];
            float p1 = q1[r] * kk1[r];
            p0 += __shfl_xor(p0, 1); p0 += __shfl_xor(p0, 2); p0 += __shfl_xor(p0, 4);
            p1 += __shfl_xor(p1, 1); p1 += __shfl_xor(p1, 2); p1 += __shfl_xor(p1, 4);
            const float o0 = __shfl_xor(p0, 8);
            const float o1 = __shfl_xor(p1, 8);
            const float l0 = (hbit ? o0 : p0) * inv_sqrt_dk;   // head 0
            const float l1 = (hbit ? p0 : o0) * inv_sqrt_dk;   // head 1
            const float l2 = (hbit ? o1 : p1) * inv_sqrt_dk;   // head 2
            const float l3 = (hbit ? p1 : o1) * inv_sqrt_dk;   // head 3
            const float mx = fmaxf(fmaxf(l0, l1), fmaxf(l2, l3));
            const float e0 = expf(l0 - mx), e1 = expf(l1 - mx);
            const float e2 = expf(l2 - mx), e3 = expf(l3 - mx);
            const float inv = 1.f / (e0 + e1 + e2 + e3);
            w0[r] = ((hbit ? e1 : e0) * inv) * vv0[r];
            w1[r] = ((hbit ? e3 : e2) * inv) * vv1[r];
        }

        // transpose C/D layout -> A layout through LDS (wave-private tile)
        float* wb = swb + wloc * 512;
#pragma unroll
        for (int r = 0; r < 8; ++r) {
            const int m = r + (half << 3);
            wb[m * 32 + col]      = w0[r];
            wb[m * 32 + col + 16] = w1[r];
        }
        __threadfence_block();
        asm volatile("s_wait_dscnt 0" ::: "memory");
        const float* myrow = wb + col * 32;
        v16bf aw;
#pragma unroll
        for (int e = 0; e < 16; ++e) aw[e] = (__bf16)myrow[c0 + e + (e & 8)];
        asm volatile("" ::: "memory");   // keep reads before next-iter LDS writes

        const v8f f0 = wmma_bf16(aw, pf0, z);
        const v8f f1 = wmma_bf16(aw, pf1, z);
#pragma unroll
        for (int r = 0; r < 8; ++r) {
            const int pt = tile * 16 + r + (half << 3);
            if (pt < npts) {
                float* orow = out + (size_t)pt * 64;
                orow[col]      += f0[r];
                orow[col + 16] += f1[r];
            }
        }
    }
}

// ---------------------------------------------------------------------------
// Host launcher
// ---------------------------------------------------------------------------
extern "C" void kernel_launch(void* const* d_in, const int* in_sizes, int n_in,
                              void* d_out, int out_size, void* d_ws, size_t ws_size,
                              hipStream_t stream) {
    (void)n_in; (void)out_size; (void)ws_size;
    const float* feats          = (const float*)d_in[0];
    const int* nbr3d            = (const int*)d_in[1];
    const unsigned char* mask3d = (const unsigned char*)d_in[2];
    const int* nbr2d            = (const int*)d_in[3];
    const unsigned char* mask2d = (const unsigned char*)d_in[4];
    const float* W3a = (const float*)d_in[5];
    const float* W3b = (const float*)d_in[6];
    const float* W2a = (const float*)d_in[7];
    const float* W2b = (const float*)d_in[8];
    const float* bng = (const float*)d_in[9];
    const float* bnb = (const float*)d_in[10];
    const float* bnm = (const float*)d_in[11];
    const float* bnv = (const float*)d_in[12];
    const float* wq    = (const float*)d_in[13];
    const float* wkv   = (const float*)d_in[14];
    const float* wproj = (const float*)d_in[15];
    const int npts = in_sizes[0] / 32;

    char* ws = (char*)d_ws;
    size_t off = 0;
    auto carve = [&](size_t bytes) {
        char* p = ws + off;
        off = (off + bytes + 255) & ~(size_t)255;
        return p;
    };
    __bf16* wf3a  = (__bf16*)carve(27 * 1024 * sizeof(__bf16));
    __bf16* wf3b  = (__bf16*)carve(27 * 1024 * sizeof(__bf16));
    __bf16* wf2a  = (__bf16*)carve(9 * 1024 * sizeof(__bf16));
    __bf16* wf2b  = (__bf16*)carve(9 * 1024 * sizeof(__bf16));
    __bf16* attnf = (__bf16*)carve(4096 * sizeof(__bf16));
    __bf16* h     = (__bf16*)carve((size_t)npts * 32 * sizeof(__bf16));
    __bf16* g     = (__bf16*)carve((size_t)npts * 32 * sizeof(__bf16));
    __bf16* f3db  = (__bf16*)carve((size_t)npts * 32 * sizeof(__bf16));
    __bf16* f2db  = (__bf16*)carve((size_t)npts * 32 * sizeof(__bf16));

    // --- weight -> bf16 B-fragment conversion ---
    prep_frag_kernel<<<(27 * 1024 + 255) / 256, 256, 0, stream>>>(W3a, wf3a, 27, 32, 0);
    prep_frag_kernel<<<(27 * 1024 + 255) / 256, 256, 0, stream>>>(W3b, wf3b, 27, 32, 0);
    prep_frag_kernel<<<(9 * 1024 + 255) / 256, 256, 0, stream>>>(W2a, wf2a, 9, 32, 0);
    prep_frag_kernel<<<(9 * 1024 + 255) / 256, 256, 0, stream>>>(W2b, wf2b, 9, 32, 0);
    prep_frag_kernel<<<4, 256, 0, stream>>>(wq,    attnf + 0,    1, 32, 0);
    prep_frag_kernel<<<4, 256, 0, stream>>>(wkv,   attnf + 1024, 1, 64, 0);
    prep_frag_kernel<<<4, 256, 0, stream>>>(wkv,   attnf + 2048, 1, 64, 32);
    prep_frag_kernel<<<4, 256, 0, stream>>>(wproj, attnf + 3072, 1, 32, 0);

    const int CB = 1024;  // 8 waves/block, grid-stride over 18750 tile pairs
    // conv1: feats(f32,32) -> h(bf16), BN layer 0
    subm_conv<true, 27, false><<<CB, 256, 0, stream>>>(
        feats, 32, nbr3d, mask3d, wf3a, bng + 0, bnb + 0, bnm + 0, bnv + 0,
        h, nullptr, npts);
    // conv2: h(bf16) -> f3d: bf16 shadow + f32 rows in d_out[:,0:32], BN layer 1
    subm_conv<false, 27, true><<<CB, 256, 0, stream>>>(
        h, 32, nbr3d, mask3d, wf3b, bng + 32, bnb + 32, bnm + 32, bnv + 32,
        f3db, (float*)d_out, npts);
    // conv3: f3d(bf16) -> g(bf16), BN layer 2
    subm_conv<false, 9, false><<<CB, 256, 0, stream>>>(
        f3db, 32, nbr2d, mask2d, wf2a, bng + 64, bnb + 64, bnm + 64, bnv + 64,
        g, nullptr, npts);
    // conv4: g(bf16) -> f2d: bf16 shadow + f32 rows in d_out[:,32:64], BN layer 3
    subm_conv<false, 9, true><<<CB, 256, 0, stream>>>(
        g, 32, nbr2d, mask2d, wf2b, bng + 96, bnb + 96, bnm + 96, bnv + 96,
        f2db, (float*)d_out + 32, npts);
    // attention + residual fuse into d_out[:,0:32]
    attn_fuse<<<512, 256, 0, stream>>>((float*)d_out, attnf, f3db, f2db, npts);
}
